// EmbeddedGaussianNonLocalBlock_36103495090206
// MI455X (gfx1250) — compile-verified, matrix operand order checked by reading, not run on hardware
//
#include <hip/hip_runtime.h>
#include <hip/hip_bf16.h>

typedef __attribute__((ext_vector_type(16))) _Float16 v16h;
typedef __attribute__((ext_vector_type(8)))  _Float16 v8h;
typedef __attribute__((ext_vector_type(8)))  float    v8f;
typedef __attribute__((ext_vector_type(4)))  unsigned int u32x4;
typedef __attribute__((ext_vector_type(8)))  int      i32x8;
typedef __attribute__((ext_vector_type(4)))  int      i32x4;

#define BATCH 8
#define CCH   256
#define EMB   128
#define NPIX  4096

// Build a 16x32 f16 WMMA operand fragment from contiguous f16 memory.
// Caller passes base + 8*half; lo covers K = 8h..8h+7, hi covers K = 16+8h..+7.
static __device__ __forceinline__ v16h frag_h(const _Float16* p) {
  v8h lo = *reinterpret_cast<const v8h*>(p);
  v8h hi = *reinterpret_cast<const v8h*>(p + 16);
  v16h r;
#pragma unroll
  for (int i = 0; i < 8; ++i) { r[i] = lo[i]; r[i + 8] = hi[i]; }
  return r;
}

// Same, converting from f32 memory (weights) on the fly.
static __device__ __forceinline__ v16h frag_f32(const float* p) {
  v16h r;
#pragma unroll
  for (int i = 0; i < 8; ++i) { r[i] = (_Float16)p[i]; r[i + 8] = (_Float16)p[i + 16]; }
  return r;
}

static __device__ __forceinline__ v8f wmma16(v16h a, v16h b, v8f c) {
  return __builtin_amdgcn_wmma_f32_16x16x32_f16(
      false, a, false, b, (short)0, c, false, false);
}

// Low 32 bits of a flat LDS address == LDS byte offset (aperture scheme).
static __device__ __forceinline__ unsigned int lds_off(const void* p) {
  return (unsigned int)(uintptr_t)p;
}

// Tensor Data Mover: async 2D tile load Global -> LDS.
// data_size = 8 bytes; dims/strides in 8-byte units; rows in elements.
// D# layout per CDNA5 ISA 8.3-8.5 (group0: ctrl/lds/global/type, group1: dims).
static __device__ __forceinline__ void tdm_load_2d(
    unsigned int ldsoff, const void* gptr,
    unsigned int tile_d0, unsigned int tile_d1,
    unsigned int tensor_d0, unsigned int tensor_d1,
    unsigned int d0_stride) {
  unsigned long long ga = (unsigned long long)(uintptr_t)gptr;
  u32x4 g0;
  g0[0] = 1u;                                   // count=1, user descriptor
  g0[1] = ldsoff;                               // lds_addr
  g0[2] = (unsigned int)ga;                     // global_addr[31:0]
  g0[3] = (unsigned int)((ga >> 32) & 0x01FFFFFFu) | (2u << 30);  // addr hi | type=2
  i32x8 g1;
  g1[0] = (int)(3u << 16);                      // data_size=3 (8B), no multicast
  g1[1] = (int)(tensor_d0 << 16);               // tensor_dim0 [79:48] low half
  g1[2] = (int)((tensor_d0 >> 16) | (tensor_d1 << 16));   // dim0 hi | dim1 lo
  g1[3] = (int)((tensor_d1 >> 16) | (tile_d0 << 16));     // dim1 hi | tile_dim0
  g1[4] = (int)(tile_d1 & 0xFFFFu);             // tile_dim1 | tile_dim2=0
  g1[5] = (int)d0_stride;                       // tensor_dim0_stride [191:160]
  g1[6] = 0;                                    // stride hi | dim1_stride lo
  g1[7] = 0;
  i32x4 z4 = {0, 0, 0, 0};
#if __clang_major__ >= 23
  i32x8 z8 = {0, 0, 0, 0, 0, 0, 0, 0};
  __builtin_amdgcn_tensor_load_to_lds(g0, g1, z4, z4, z8, 0);
#else
  __builtin_amdgcn_tensor_load_to_lds(g0, g1, z4, z4, 0);
#endif
}

// ---------------------------------------------------------------------------
// Kernel 1: fused q/k/v 1x1-conv projections.
// out[n,e] = sum_c x[b,c,n] * W[e,c] + bias[e]
// q,k stored row-major [N,E] f16; v stored transposed [E,N] f16.
// ---------------------------------------------------------------------------
__global__ __launch_bounds__(32) void qkv_proj_kernel(
    const float* __restrict__ x,
    const float* __restrict__ wq, const float* __restrict__ bq,
    const float* __restrict__ wk, const float* __restrict__ bk,
    const float* __restrict__ wv, const float* __restrict__ bv,
    _Float16* __restrict__ qo, _Float16* __restrict__ ko,
    _Float16* __restrict__ vto) {
  const int lane = threadIdx.x & 31;
  const int mrow = lane & 15;
  const int hf   = lane >> 4;
  const int n0   = blockIdx.x * 16;
  const int proj = blockIdx.y >> 3;           // 0=q, 1=k, 2=v
  const int e0   = (blockIdx.y & 7) * 16;
  const int b    = blockIdx.z;

  const float* w  = (proj == 0) ? wq : (proj == 1) ? wk : wv;
  const float* bi = (proj == 0) ? bq : (proj == 1) ? bk : bv;

  const int n = n0 + mrow;     // A-operand row (pixel)
  const int e = e0 + mrow;     // B-operand column (embed) == store column

  v8f acc = {};
#pragma unroll
  for (int kc = 0; kc < CCH; kc += 32) {
    // A: x is channels-major -> strided gather along c, convert to f16
    const float* xa = x + ((size_t)b * CCH + kc + 8 * hf) * NPIX + n;
    v16h a;
#pragma unroll
    for (int j = 0; j < 8; ++j) {
      a[j]     = (_Float16)xa[(size_t)j * NPIX];
      a[j + 8] = (_Float16)xa[(size_t)(16 + j) * NPIX];
    }
    v16h bf = frag_f32(w + (size_t)e * CCH + kc + 8 * hf);
    acc = wmma16(a, bf, acc);
  }

  const float bias = bi[e];
  if (proj < 2) {
    _Float16* dst = (proj == 0 ? qo : ko) + (size_t)b * NPIX * EMB;
#pragma unroll
    for (int v = 0; v < 8; ++v)
      dst[(size_t)(n0 + 8 * hf + v) * EMB + e] = (_Float16)(acc[v] + bias);
  } else {
    _Float16* dst = vto + (size_t)b * EMB * NPIX + (size_t)e * NPIX;
#pragma unroll
    for (int v = 0; v < 8; ++v)
      dst[n0 + 8 * hf + v] = (_Float16)(acc[v] + bias);
  }
}

// ---------------------------------------------------------------------------
// Kernel 2: flash attention with TDM-staged K/V in LDS (double buffered).
// WG = 4 waves = 64 query rows; wave 0 drives the Tensor Data Mover.
// Per 32-key block: 8 wmma (S) + online softmax + 8 wmma (PV).
// ---------------------------------------------------------------------------
__global__ __launch_bounds__(128) void attn_kernel(
    const _Float16* __restrict__ q, const _Float16* __restrict__ k,
    const _Float16* __restrict__ vt, _Float16* __restrict__ feat) {
  __shared__ __align__(16) _Float16 ldsK[2][32][EMB];   // 2 x 8 KB: K rows
  __shared__ __align__(16) _Float16 ldsV[2][EMB][32];   // 2 x 8 KB: V^T rows
  __shared__ __align__(16) _Float16 ldsP[4][16][32];    // per-wave P tile
  const int lane = threadIdx.x & 31;
  const int wid  = threadIdx.x >> 5;
  const int mrow = lane & 15;
  const int hf   = lane >> 4;
  const int col  = mrow;
  const int b    = blockIdx.y;
  const int n0   = blockIdx.x * 64 + wid * 16;

  const _Float16* kbase = k + (size_t)b * NPIX * EMB;   // [N][E]
  const _Float16* vbase = vt + (size_t)b * EMB * NPIX;  // [E][N]

  // Persistent Q fragments: 16 rows x 128 e
  v16h qf[4];
  const _Float16* qrow = q + ((size_t)b * NPIX + n0 + mrow) * EMB;
#pragma unroll
  for (int t = 0; t < 4; ++t) qf[t] = frag_h(qrow + 32 * t + 8 * hf);

  v8f o[8] = {};
  float mx[8], ls[8];
#pragma unroll
  for (int v = 0; v < 8; ++v) { mx[v] = -1e30f; ls[v] = 0.0f; }

  // Prologue: stage key block 0 into buffer 0.
  if (threadIdx.x < 32) {
    // K tile: 32 rows x 256B;  V tile: 128 rows x 64B (from vT)
    tdm_load_2d(lds_off(&ldsK[0][0][0]), kbase, 32, 32, 32, NPIX, 32);
    tdm_load_2d(lds_off(&ldsV[0][0][0]), vbase, 8, EMB, NPIX / 4, EMB, NPIX / 4);
  }

  int cur = 0;
  for (int kb = 0; kb < NPIX; kb += 32) {
    __syncthreads();   // all waves done reading buffer cur^1 (prev iteration)
    if (threadIdx.x < 32) {
      if (kb + 32 < NPIX) {
        tdm_load_2d(lds_off(&ldsK[cur ^ 1][0][0]), kbase + (size_t)(kb + 32) * EMB,
                    32, 32, 32, NPIX, 32);
        tdm_load_2d(lds_off(&ldsV[cur ^ 1][0][0]), vbase + kb + 32,
                    8, EMB, NPIX / 4, EMB, NPIX / 4);
        __builtin_amdgcn_s_wait_tensorcnt(2);  // in-order: block kb's 2 ops done
      } else {
        __builtin_amdgcn_s_wait_tensorcnt(0);
      }
    }
    __syncthreads();   // buffer cur now valid for all waves

    // ---- S tiles: [16 q-rows] x [32 keys], K from LDS ----
    v8f s0 = {}, s1 = {};
#pragma unroll
    for (int t = 0; t < 4; ++t) {
      s0 = wmma16(qf[t], frag_h(&ldsK[cur][col][32 * t + 8 * hf]), s0);
      s1 = wmma16(qf[t], frag_h(&ldsK[cur][16 + col][32 * t + 8 * hf]), s1);
    }

    // ---- online softmax; row r = v + 8*hf spans 16 lanes of this half ----
#pragma unroll
    for (int v = 0; v < 8; ++v) {
      float t = fmaxf(s0[v], s1[v]);
#pragma unroll
      for (int msk = 1; msk <= 8; msk <<= 1) t = fmaxf(t, __shfl_xor(t, msk, 32));
      const float mn = fmaxf(mx[v], t);
      const float sc = __expf(mx[v] - mn);
      const float p0 = __expf(s0[v] - mn);
      const float p1 = __expf(s1[v] - mn);
      float rs = p0 + p1;
#pragma unroll
      for (int msk = 1; msk <= 8; msk <<= 1) rs += __shfl_xor(rs, msk, 32);
      ls[v] = ls[v] * sc + rs;
      mx[v] = mn;
#pragma unroll
      for (int et = 0; et < 8; ++et) o[et][v] *= sc;   // rescale O rows
      ldsP[wid][8 * hf + v][col]      = (_Float16)p0;
      ldsP[wid][8 * hf + v][col + 16] = (_Float16)p1;
    }
    asm volatile("s_wait_dscnt 0" ::: "memory");       // P stores visible (same wave)

    // ---- O += P V ;  P as A-operand from LDS, V^T columns from LDS ----
    v16h pa = frag_h(&ldsP[wid][mrow][8 * hf]);
#pragma unroll
    for (int et = 0; et < 8; ++et)
      o[et] = wmma16(pa, frag_h(&ldsV[cur][et * 16 + col][8 * hf]), o[et]);

    cur ^= 1;
  }

  // ---- normalize and store feat[b, n, e] (f16, row-major) ----
  _Float16* frow = feat + ((size_t)b * NPIX + n0 + 8 * hf) * EMB;
#pragma unroll
  for (int v = 0; v < 8; ++v) {
    const float inv = 1.0f / ls[v];
#pragma unroll
    for (int et = 0; et < 8; ++et)
      frow[(size_t)v * EMB + et * 16 + col] = (_Float16)(o[et][v] * inv);
  }
}

// ---------------------------------------------------------------------------
// Kernel 3: output 1x1 conv + bias + residual, back to channels-major f32.
// ---------------------------------------------------------------------------
__global__ __launch_bounds__(32) void out_proj_kernel(
    const _Float16* __restrict__ feat, const float* __restrict__ wo,
    const float* __restrict__ bo, const float* __restrict__ x,
    float* __restrict__ out) {
  const int lane = threadIdx.x & 31;
  const int mrow = lane & 15;
  const int hf   = lane >> 4;
  const int n0   = blockIdx.x * 16;
  const int c0   = blockIdx.y * 16;
  const int b    = blockIdx.z;
  const int c    = c0 + mrow;

  v8f acc = {};
  const _Float16* fr = feat + ((size_t)b * NPIX + n0 + mrow) * EMB;
  const float*    wr = wo + (size_t)c * EMB;
#pragma unroll
  for (int kc = 0; kc < EMB; kc += 32)
    acc = wmma16(frag_h(fr + kc + 8 * hf), frag_f32(wr + kc + 8 * hf), acc);

  const float bias = bo[c];
  const size_t base = ((size_t)b * CCH + c) * NPIX + n0 + 8 * hf;  // 8 consecutive n
  const float4* xi = reinterpret_cast<const float4*>(x + base);
  float4*       oo = reinterpret_cast<float4*>(out + base);
  float4 r0 = xi[0], r1 = xi[1];
  float4 w0, w1;
  w0.x = acc[0] + bias + r0.x;  w0.y = acc[1] + bias + r0.y;
  w0.z = acc[2] + bias + r0.z;  w0.w = acc[3] + bias + r0.w;
  w1.x = acc[4] + bias + r1.x;  w1.y = acc[5] + bias + r1.y;
  w1.z = acc[6] + bias + r1.z;  w1.w = acc[7] + bias + r1.w;
  oo[0] = w0;
  oo[1] = w1;
}

// ---------------------------------------------------------------------------
extern "C" void kernel_launch(void* const* d_in, const int* in_sizes, int n_in,
                              void* d_out, int out_size, void* d_ws, size_t ws_size,
                              hipStream_t stream) {
  (void)in_sizes; (void)n_in; (void)out_size; (void)ws_size;
  const float* x  = (const float*)d_in[0];
  const float* wq = (const float*)d_in[1];
  const float* bq = (const float*)d_in[2];
  const float* wk = (const float*)d_in[3];
  const float* bk = (const float*)d_in[4];
  const float* wv = (const float*)d_in[5];
  const float* bv = (const float*)d_in[6];
  const float* wo = (const float*)d_in[7];
  const float* bo = (const float*)d_in[8];

  // Workspace: q[N,E], k[N,E], vT[E,N], feat[N,E] per batch, all f16 (32 MB).
  _Float16* ws = (_Float16*)d_ws;
  const size_t sz = (size_t)BATCH * NPIX * EMB;
  _Float16* q  = ws;
  _Float16* kk = ws + sz;
  _Float16* vt = ws + 2 * sz;
  _Float16* ft = ws + 3 * sz;

  qkv_proj_kernel<<<dim3(NPIX / 16, 24, BATCH), 32, 0, stream>>>(
      x, wq, bq, wk, bk, wv, bv, q, kk, vt);
  attn_kernel<<<dim3(NPIX / 64, BATCH), 128, 0, stream>>>(q, kk, vt, ft);
  out_proj_kernel<<<dim3(NPIX / 16, CCH / 16, BATCH), 32, 0, stream>>>(
      ft, wo, bo, x, (float*)d_out);
}
